// DenseDCLLlayer_68977174773809
// MI455X (gfx1250) — matrix operand order repaired and checked by reading, not compiled, not run on hardware
//
#include <hip/hip_runtime.h>

// ---------------------------------------------------------------------------
// DenseDCLL layer on MI455X (gfx1250, wave32, WMMA, async LDS copies)
// isyn = 0.85*prev_isyn + x@W.T ; vmem = 0.9*prev_vmem + isyn
// eps0 = 0.85*prev_eps0 + x     ; eps1 = 0.9*prev_eps1 + eps0
// pv   = eps1@W.T ; spike = vmem>0.5 ; head = log_softmax(sigmoid(pv@Wo.T+bo))
// ---------------------------------------------------------------------------

#define B_DIM   4096
#define K_DIM   4096   // D_IN
#define N_DIM   4096   // D_OUT
#define D_CLS   10
#define NN      ((size_t)B_DIM * (size_t)N_DIM)

#define ALPHA   0.9f
#define ALPHAS  0.85f

typedef __bf16         v16bf __attribute__((ext_vector_type(16)));
typedef float          v8f   __attribute__((ext_vector_type(8)));
typedef unsigned int   u32x4 __attribute__((ext_vector_type(4)));
typedef float          f32x4 __attribute__((ext_vector_type(4)));
typedef unsigned short u16x4 __attribute__((ext_vector_type(4)));

union FragAB { v16bf v; u32x4 q[2]; };

__device__ __forceinline__ unsigned short f2bf(float f) {
    unsigned int u = __float_as_uint(f);
    unsigned int r = (u + 0x7FFFu + ((u >> 16) & 1u)) >> 16; // RNE to bf16
    return (unsigned short)r;
}

// CDNA5 async global->LDS copy, 16B per lane, tracked by ASYNCcnt.
// LDS operand is the 32-bit LDS offset = low 32 bits of the generic pointer
// (shared aperture has zero low bits, ISA 10.2).
__device__ __forceinline__ void async_cp16(void* lds, const void* gptr) {
    unsigned l = (unsigned)(unsigned long long)lds;
    asm volatile("global_load_async_to_lds_b128 %0, %1, off"
                 :: "v"(l), "v"(gptr) : "memory");
}
__device__ __forceinline__ void wait_async0() {
    asm volatile("s_wait_asynccnt 0x0" ::: "memory");
}

// ---------------------------------------------------------------------------
// Kernel 1: eps updates + bf16 casts of x and eps1 (4 elems/thread)
// ---------------------------------------------------------------------------
__global__ __launch_bounds__(256) void eps_update_kernel(
    const float* __restrict__ x, const float* __restrict__ pe0,
    const float* __restrict__ pe1,
    float* __restrict__ out_e0, float* __restrict__ out_e1,
    unsigned short* __restrict__ xbf, unsigned short* __restrict__ e1bf)
{
    size_t i = ((size_t)blockIdx.x * blockDim.x + threadIdx.x) * 4;
    f32x4 xv  = *(const f32x4*)&x[i];
    f32x4 e0v = *(const f32x4*)&pe0[i];
    f32x4 e1v = *(const f32x4*)&pe1[i];
    f32x4 e0, e1;
    u16x4 xb, eb;
#pragma unroll
    for (int c = 0; c < 4; ++c) {
        float e0c = ALPHAS * e0v[c] + xv[c];
        float e1c = ALPHA  * e1v[c] + e0c;
        e0[c] = e0c; e1[c] = e1c;
        xb[c] = f2bf(xv[c]); eb[c] = f2bf(e1c);
    }
    *(f32x4*)&out_e0[i] = e0;
    *(f32x4*)&out_e1[i] = e1;
    *(u16x4*)&xbf[i]  = xb;
    *(u16x4*)&e1bf[i] = eb;
}

// ---------------------------------------------------------------------------
// Kernel 2: W fp32 -> bf16
// ---------------------------------------------------------------------------
__global__ __launch_bounds__(256) void wcast_kernel(
    const float* __restrict__ W, unsigned short* __restrict__ wbf)
{
    size_t i = ((size_t)blockIdx.x * blockDim.x + threadIdx.x) * 4;
    f32x4 wv = *(const f32x4*)&W[i];
    u16x4 wb;
#pragma unroll
    for (int c = 0; c < 4; ++c) wb[c] = f2bf(wv[c]);
    *(u16x4*)&wbf[i] = wb;
}

// ---------------------------------------------------------------------------
// Kernel 3: fused dual GEMM (x@W.T and eps1@W.T share W tiles) + epilogue
// WG: 256 thr = 8 wave32 waves (4 M x 2 N); WG tile 128(M) x 64(N)
// Wave tile: 32x32 = 2x2 WMMA tiles, dual accumulators.
// Double-buffered LDS fed by global_load_async_to_lds_b128; 1 barrier/iter.
// ---------------------------------------------------------------------------
#define TK 32
#define ST 40   // padded LDS row stride in halves (80B: 16B aligned, skewed)

__global__ __launch_bounds__(256) void dual_gemm_kernel(
    const unsigned short* __restrict__ Xbf,
    const unsigned short* __restrict__ E1bf,
    const unsigned short* __restrict__ Wbf,
    const float* __restrict__ prev_isyn, const float* __restrict__ prev_vmem,
    float* __restrict__ out_isyn, float* __restrict__ out_vmem,
    float* __restrict__ out_spk,  float* __restrict__ pv_ws)
{
    __shared__ __align__(16) unsigned short sX[2][128 * ST];
    __shared__ __align__(16) unsigned short sE[2][128 * ST];
    __shared__ __align__(16) unsigned short sW[2][64 * ST];

    const int tid  = threadIdx.x;
    const int lane = tid & 31;
    const int w    = tid >> 5;
    const int wm   = w >> 1;        // 0..3  (M waves)
    const int wn   = w & 1;         // 0..1  (N waves)
    const int bm   = blockIdx.y * 128;
    const int bn   = blockIdx.x * 64;

    v8f accI[2][2] = {};            // x@W.T
    v8f accP[2][2] = {};            // eps1@W.T

    const int lrow = lane & 15;
    const int kA   = (lane < 16) ? 0 : 8;   // A frag K 0-7/16-23 vs 8-15/24-31
    const int kB   = (lane < 16) ? 0 : 8;   // B frag K 0-7/16-23 vs 8-15/24-31

    // per-thread staging coordinates (5 x 16B async copies per K-step)
    const int xr0 = tid >> 2,          xc0 = (tid & 3) * 8;          // chunk tid
    const int xr1 = (tid + 256) >> 2,  xc1 = ((tid + 256) & 3) * 8;  // chunk tid+256
    const int wr  = tid >> 2,          wc  = (tid & 3) * 8;

    auto stage = [&](int buf, int kk) {
        size_t gx0 = (size_t)(bm + xr0) * K_DIM + kk + xc0;
        size_t gx1 = (size_t)(bm + xr1) * K_DIM + kk + xc1;
        size_t gw  = (size_t)(bn + wr)  * K_DIM + kk + wc;
        async_cp16(&sX[buf][xr0 * ST + xc0], &Xbf[gx0]);
        async_cp16(&sX[buf][xr1 * ST + xc1], &Xbf[gx1]);
        async_cp16(&sE[buf][xr0 * ST + xc0], &E1bf[gx0]);
        async_cp16(&sE[buf][xr1 * ST + xc1], &E1bf[gx1]);
        async_cp16(&sW[buf][wr * ST + wc],   &Wbf[gw]);
    };

    stage(0, 0);                     // prologue: prefetch first K-slab

    for (int kk = 0; kk < K_DIM; kk += TK) {
        const int cur = (kk >> 5) & 1;
        wait_async0();               // own async copies into `cur` done
        __syncthreads();             // everyone's copies done / reads drained
        if (kk + TK < K_DIM)
            stage(cur ^ 1, kk + TK); // overlap next slab fetch with compute

        // ---- B fragments from sW (shared by both GEMMs) ----
        FragAB bfrag[2];
#pragma unroll
        for (int nt = 0; nt < 2; ++nt) {
            int r = wn * 32 + nt * 16 + lrow;
            bfrag[nt].q[0] = *(const u32x4*)&sW[cur][r * ST + kB];
            bfrag[nt].q[1] = *(const u32x4*)&sW[cur][r * ST + kB + 16];
        }
        // ---- A fragments + WMMA ----
#pragma unroll
        for (int mt = 0; mt < 2; ++mt) {
            int r = wm * 32 + mt * 16 + lrow;
            FragAB ax, ae;
            ax.q[0] = *(const u32x4*)&sX[cur][r * ST + kA];
            ax.q[1] = *(const u32x4*)&sX[cur][r * ST + kA + 16];
            ae.q[0] = *(const u32x4*)&sE[cur][r * ST + kA];
            ae.q[1] = *(const u32x4*)&sE[cur][r * ST + kA + 16];
#pragma unroll
            for (int nt = 0; nt < 2; ++nt) {
                accI[mt][nt] = __builtin_amdgcn_wmma_f32_16x16x32_bf16(
                    false, ax.v, false, bfrag[nt].v, (short)0, accI[mt][nt], false, false);
                accP[mt][nt] = __builtin_amdgcn_wmma_f32_16x16x32_bf16(
                    false, ae.v, false, bfrag[nt].v, (short)0, accP[mt][nt], false, false);
            }
        }
    }

    // ---- epilogue: leaky integrator fusion + spike + pv to workspace ----
#pragma unroll
    for (int mt = 0; mt < 2; ++mt) {
#pragma unroll
        for (int nt = 0; nt < 2; ++nt) {
            int gm0 = bm + wm * 32 + mt * 16 + ((lane < 16) ? 0 : 8);
            int gn  = bn + wn * 32 + nt * 16 + (lane & 15);
#pragma unroll
            for (int r = 0; r < 8; ++r) {
                size_t idx = (size_t)(gm0 + r) * N_DIM + gn;
                float is = ALPHAS * prev_isyn[idx] + accI[mt][nt][r];
                float vm = ALPHA  * prev_vmem[idx] + is;
                out_isyn[idx] = is;
                out_vmem[idx] = vm;
                out_spk[idx]  = (vm > 0.5f) ? 1.0f : 0.0f;
                pv_ws[idx]    = accP[mt][nt][r];
            }
        }
    }
}

// ---------------------------------------------------------------------------
// Kernel 4: classifier head, one block per batch row.
// ---------------------------------------------------------------------------
__global__ __launch_bounds__(256) void head_kernel(
    const float* __restrict__ pv, const float* __restrict__ Wo,
    const float* __restrict__ bo, float* __restrict__ outp)
{
    __shared__ float sred[256];
    __shared__ float z[D_CLS];
    const int row = blockIdx.x;
    const int tid = threadIdx.x;

    float part[D_CLS];
#pragma unroll
    for (int c = 0; c < D_CLS; ++c) part[c] = 0.0f;

    for (int o = tid; o < N_DIM; o += 256) {
        float p = pv[(size_t)row * N_DIM + o];
#pragma unroll
        for (int c = 0; c < D_CLS; ++c)
            part[c] += p * Wo[(size_t)c * N_DIM + o];
    }
    for (int c = 0; c < D_CLS; ++c) {
        sred[tid] = part[c];
        __syncthreads();
        for (int s = 128; s > 0; s >>= 1) {
            if (tid < s) sred[tid] += sred[tid + s];
            __syncthreads();
        }
        if (tid == 0) z[c] = sred[0];
        __syncthreads();
    }
    if (tid == 0) {
        float s[D_CLS], mx = -1e30f, sum = 0.0f;
#pragma unroll
        for (int c = 0; c < D_CLS; ++c) {
            s[c] = 1.0f / (1.0f + __expf(-(z[c] + bo[c])));
            mx = fmaxf(mx, s[c]);
        }
#pragma unroll
        for (int c = 0; c < D_CLS; ++c) sum += __expf(s[c] - mx);
        float lse = mx + __logf(sum);
#pragma unroll
        for (int c = 0; c < D_CLS; ++c)
            outp[(size_t)row * D_CLS + c] = s[c] - lse;
    }
}

// ---------------------------------------------------------------------------
extern "C" void kernel_launch(void* const* d_in, const int* in_sizes, int n_in,
                              void* d_out, int out_size, void* d_ws, size_t ws_size,
                              hipStream_t stream)
{
    const float* x    = (const float*)d_in[0];
    const float* pisy = (const float*)d_in[1];
    const float* pvme = (const float*)d_in[2];
    const float* pe0  = (const float*)d_in[3];
    const float* pe1  = (const float*)d_in[4];
    const float* W    = (const float*)d_in[5];
    const float* Wo   = (const float*)d_in[6];
    const float* bo   = (const float*)d_in[7];

    float* out = (float*)d_out;
    float* o_isyn = out;
    float* o_vmem = out + NN;
    float* o_eps0 = out + 2 * NN;
    float* o_eps1 = out + 3 * NN;
    float* o_spk  = out + 4 * NN;
    float* o_pvo  = out + 5 * NN;

    // workspace: xbf | e1bf | wbf (bf16) | pv (f32)
    unsigned short* xbf  = (unsigned short*)d_ws;
    unsigned short* e1bf = xbf + NN;
    unsigned short* wbf  = e1bf + NN;
    float*          pv   = (float*)(wbf + NN);

    eps_update_kernel<<<NN / (256 * 4), 256, 0, stream>>>(
        x, pe0, pe1, o_eps0, o_eps1, xbf, e1bf);

    wcast_kernel<<<NN / (256 * 4), 256, 0, stream>>>(W, wbf);

    dim3 grid(N_DIM / 64, B_DIM / 128, 1);
    dual_gemm_kernel<<<grid, 256, 0, stream>>>(
        xbf, e1bf, wbf, pisy, pvme, o_isyn, o_vmem, o_spk, pv);

    head_kernel<<<B_DIM, 256, 0, stream>>>(pv, Wo, bo, o_pvo);
}
